// LSTMPositionPredictor_79577154060668
// MI455X (gfx1250) — compile-verified, matrix operand order checked by reading, not run on hardware
//
#include <hip/hip_runtime.h>
#include <hip/hip_bf16.h>
#include <math.h>

// Problem constants (match reference)
#define Hh   2048
#define G4H  8192      // 4*H
#define Bsz  64
#define Tlen 128
#define TGTn 64

typedef unsigned int u32x4 __attribute__((ext_vector_type(4)));
typedef __bf16 bf16x16 __attribute__((ext_vector_type(16)));
typedef float f32x8 __attribute__((ext_vector_type(8)));

union Frag { bf16x16 v; u32x4 u[2]; };

__device__ __forceinline__ unsigned short f2bf(float f) {
  unsigned int u = __float_as_uint(f);
  u += 0x7FFFu + ((u >> 16) & 1u);        // round-to-nearest-even
  return (unsigned short)(u >> 16);
}

__device__ __forceinline__ float sigmoidf_(float x) {
  return 1.0f / (1.0f + expf(-x));
}

// Accumulate a 16x16 f32 tile: D += A(16x2048 rows at Arow) x W^T(cols at Wrow).
// A row-major (MxK), W row-major (G4H x K) so W row g is the K-contiguous column g of B.
__device__ __forceinline__ f32x8 gemm_rowcol(f32x8 acc,
    const unsigned short* __restrict__ Arow,
    const unsigned short* __restrict__ Wrow,
    int basek) {
  #pragma unroll 4
  for (int kk = 0; kk < Hh; kk += 32) {
    Frag a, b;
    const u32x4* pa = (const u32x4*)(Arow + kk + basek);
    a.u[0] = pa[0];          // K = basek .. basek+7
    a.u[1] = pa[2];          // K = basek+16 .. basek+23
    const u32x4* pb = (const u32x4*)(Wrow + kk + basek);
    b.u[0] = pb[0];
    b.u[1] = pb[2];
    acc = __builtin_amdgcn_wmma_f32_16x16x32_bf16(
        false, a.v, false, b.v, (short)0, acc, false, false);
  }
  return acc;
}

// One LSTM step: gates = A1@W1^T (+ A2@W2^T) + bias (+ x_t@encWih^T), then
// pointwise cell update with optional length-mask freeze (packed-seq semantics).
// Block: 512 thr = 16 waves; wave (mtile = w&3, gate = w>>2); gridDim.x = H/16.
__global__ __launch_bounds__(512)
void lstm_step_kernel(const unsigned short* __restrict__ A1,
                      const unsigned short* __restrict__ W1,
                      const unsigned short* __restrict__ A2,
                      const unsigned short* __restrict__ W2,
                      const float* __restrict__ bias,     // 4H combined bih+bhh
                      const float* __restrict__ x,        // (B,T,4) or null
                      const float* __restrict__ encWih,   // (4H,4) fp32 or null
                      const int*   __restrict__ lengths,  // (B,) or null
                      int t,
                      float* __restrict__ h,
                      float* __restrict__ c,
                      unsigned short* __restrict__ hbOut) {
  __shared__ float glds[4 * 64 * 16];  // [gate][b][jj] 16 KB
  const int tid   = threadIdx.x;
  const int lane  = tid & 31;
  const int wave  = tid >> 5;
  const int mtile = wave & 3;
  const int gate  = wave >> 2;
  const int j0    = blockIdx.x * 16;
  const int m0    = mtile * 16;
  const int basek = (lane < 16) ? 0 : 8;
  const int row   = m0 + (lane & 15);
  const int g     = gate * Hh + j0 + (lane & 15);   // global gate-row of W

  f32x8 acc = {0.f,0.f,0.f,0.f,0.f,0.f,0.f,0.f};
  acc = gemm_rowcol(acc, A1 + (size_t)row * Hh, W1 + (size_t)g * Hh, basek);
  if (A2)
    acc = gemm_rowcol(acc, A2 + (size_t)row * Hh, W2 + (size_t)g * Hh, basek);

  // D layout: VGPR i -> M = i (+8 for lanes>=16), N = lane&15
  const int n  = lane & 15;
  const int mh = (lane >> 4) << 3;
  #pragma unroll
  for (int i = 0; i < 8; ++i)
    glds[(gate * 64 + m0 + mh + i) * 16 + n] = acc[i];

  __syncthreads();

  #pragma unroll
  for (int r = 0; r < 2; ++r) {
    int cidx = tid + r * 512;          // 1024 cells = 64 b x 16 jj
    int b  = cidx >> 4;
    int jj = cidx & 15;
    int j  = j0 + jj;
    float gv0 = glds[(0 * 64 + b) * 16 + jj] + bias[0 * Hh + j];
    float gv1 = glds[(1 * 64 + b) * 16 + jj] + bias[1 * Hh + j];
    float gv2 = glds[(2 * 64 + b) * 16 + jj] + bias[2 * Hh + j];
    float gv3 = glds[(3 * 64 + b) * 16 + jj] + bias[3 * Hh + j];
    if (x) {  // encoder: fuse x_t @ Wih^T (K=4) in fp32
      const float* xb = x + ((size_t)b * Tlen + t) * 4;
      float x0 = xb[0], x1 = xb[1], x2 = xb[2], x3 = xb[3];
      const float* w0 = encWih + (size_t)(0 * Hh + j) * 4;
      const float* w1 = encWih + (size_t)(1 * Hh + j) * 4;
      const float* w2 = encWih + (size_t)(2 * Hh + j) * 4;
      const float* w3 = encWih + (size_t)(3 * Hh + j) * 4;
      gv0 += w0[0]*x0 + w0[1]*x1 + w0[2]*x2 + w0[3]*x3;
      gv1 += w1[0]*x0 + w1[1]*x1 + w1[2]*x2 + w1[3]*x3;
      gv2 += w2[0]*x0 + w2[1]*x1 + w2[2]*x2 + w2[3]*x3;
      gv3 += w3[0]*x0 + w3[1]*x1 + w3[2]*x2 + w3[3]*x3;
    }
    float iv = sigmoidf_(gv0);
    float fv = sigmoidf_(gv1);
    float gg = tanhf(gv2);
    float ov = sigmoidf_(gv3);
    size_t idx = (size_t)b * Hh + j;
    float c_old = c[idx];
    float h_old = h[idx];
    float c_new = fv * c_old + iv * gg;
    float h_new = ov * tanhf(c_new);
    bool keep = lengths ? (t < lengths[b]) : true;
    float hf = keep ? h_new : h_old;
    float cf = keep ? c_new : c_old;
    h[idx] = hf;
    c[idx] = cf;
    hbOut[idx] = f2bf(hf);   // double-buffered bf16 activation for next GEMM
  }
}

__global__ void convert_kernel(const float* __restrict__ encWhh,
                               const float* __restrict__ decWih,
                               const float* __restrict__ decWhh,
                               const float* __restrict__ ebih, const float* __restrict__ ebhh,
                               const float* __restrict__ dbih, const float* __restrict__ dbhh,
                               unsigned short* __restrict__ wEnc,
                               unsigned short* __restrict__ wDecIh,
                               unsigned short* __restrict__ wDecHh,
                               float* __restrict__ biasEnc,
                               float* __restrict__ biasDec) {
  size_t n = (size_t)G4H * Hh;
  size_t stride = (size_t)gridDim.x * blockDim.x;
  for (size_t i = (size_t)blockIdx.x * blockDim.x + threadIdx.x; i < n; i += stride) {
    wEnc[i]   = f2bf(encWhh[i]);
    wDecIh[i] = f2bf(decWih[i]);
    wDecHh[i] = f2bf(decWhh[i]);
    if (i < G4H) {
      biasEnc[i] = ebih[i] + ebhh[i];
      biasDec[i] = dbih[i] + dbhh[i];
    }
  }
}

__global__ void init_kernel(float* h, float* c, unsigned short* hb0, unsigned short* hb1) {
  size_t n = (size_t)Bsz * Hh;
  size_t stride = (size_t)gridDim.x * blockDim.x;
  for (size_t i = (size_t)blockIdx.x * blockDim.x + threadIdx.x; i < n; i += stride) {
    h[i] = 0.f; c[i] = 0.f; hb0[i] = 0; hb1[i] = 0;
  }
}

__global__ void x0_kernel(const float* __restrict__ x, const int* __restrict__ lengths,
                          float* __restrict__ decx) {
  int i = threadIdx.x;
  if (i < Bsz * 4) {
    int b = i >> 2, d = i & 3;
    int L = lengths[b];
    decx[i] = x[((size_t)b * Tlen + (L - 1)) * 4 + d];
  }
}

__global__ void emb_kernel(const float* __restrict__ decx,
                           const float* __restrict__ embW,   // (H,4) fp32
                           const float* __restrict__ embb,
                           unsigned short* __restrict__ emb) {
  size_t n = (size_t)Bsz * Hh;
  size_t stride = (size_t)gridDim.x * blockDim.x;
  for (size_t i = (size_t)blockIdx.x * blockDim.x + threadIdx.x; i < n; i += stride) {
    int b = (int)(i / Hh), j = (int)(i % Hh);
    const float* w  = embW + (size_t)j * 4;
    const float* xb = decx + b * 4;
    float v = embb[j] + w[0]*xb[0] + w[1]*xb[1] + w[2]*xb[2] + w[3]*xb[3];
    emb[i] = f2bf(v);
  }
}

// fc head: one wave per (b,d) of the 64x4 output; wave32 shuffle reduction.
__global__ __launch_bounds__(256)
void fc_kernel(const float* __restrict__ h,
               const float* __restrict__ fcW,  // (4,H) fp32
               const float* __restrict__ fcb,
               float* __restrict__ outbuf, float* __restrict__ decx, int s) {
  int wv   = (int)((blockIdx.x * blockDim.x + threadIdx.x) >> 5);
  int lane = threadIdx.x & 31;
  int b = wv >> 2, d = wv & 3;
  const float* hr = h   + (size_t)b * Hh;
  const float* wr = fcW + (size_t)d * Hh;
  float acc = 0.f;
  for (int j = lane; j < Hh; j += 32) acc += hr[j] * wr[j];
  #pragma unroll
  for (int off = 16; off > 0; off >>= 1) acc += __shfl_down(acc, off, 32);
  if (lane == 0) {
    float v = acc + fcb[d];
    outbuf[((size_t)b * TGTn + s) * 4 + d]  = v;                 // predictions
    outbuf[(size_t)Bsz * TGTn * 4 + b * 4 + d] = v;              // 'out' (last wins)
    decx[b * 4 + d] = v;                                         // feedback
  }
}

__global__ void final_copy_kernel(const float* __restrict__ h, const float* __restrict__ c,
                                  float* __restrict__ outbuf) {
  size_t base = (size_t)Bsz * TGTn * 4 + (size_t)Bsz * 4;  // predictions + out
  size_t n = (size_t)Bsz * Hh;
  size_t stride = (size_t)gridDim.x * blockDim.x;
  for (size_t i = (size_t)blockIdx.x * blockDim.x + threadIdx.x; i < n; i += stride) {
    outbuf[base + i]     = h[i];
    outbuf[base + n + i] = c[i];
  }
}

extern "C" void kernel_launch(void* const* d_in, const int* in_sizes, int n_in,
                              void* d_out, int out_size, void* d_ws, size_t ws_size,
                              hipStream_t stream) {
  (void)in_sizes; (void)n_in; (void)out_size; (void)ws_size;
  const float* x       = (const float*)d_in[0];
  const int*   lengths = (const int*)  d_in[1];
  // d_in[2] = target_len (compile-time TGTn)
  const float* encWih  = (const float*)d_in[3];
  const float* encWhh  = (const float*)d_in[4];
  const float* ebih    = (const float*)d_in[5];
  const float* ebhh    = (const float*)d_in[6];
  const float* embW    = (const float*)d_in[7];
  const float* embb    = (const float*)d_in[8];
  const float* decWih  = (const float*)d_in[9];
  const float* decWhh  = (const float*)d_in[10];
  const float* dbih    = (const float*)d_in[11];
  const float* dbhh    = (const float*)d_in[12];
  const float* fcW     = (const float*)d_in[13];
  const float* fcb     = (const float*)d_in[14];
  float* outbuf = (float*)d_out;

  char* ws = (char*)d_ws;
  const size_t WB = (size_t)G4H * Hh * sizeof(unsigned short);   // 32 MB each
  unsigned short* wEnc    = (unsigned short*)(ws);
  unsigned short* wDecIh  = (unsigned short*)(ws + WB);
  unsigned short* wDecHh  = (unsigned short*)(ws + 2 * WB);
  float* biasEnc = (float*)(ws + 3 * WB);
  float* biasDec = biasEnc + G4H;
  float* h       = biasDec + G4H;
  float* c       = h + (size_t)Bsz * Hh;
  unsigned short* hb0 = (unsigned short*)(c + (size_t)Bsz * Hh);
  unsigned short* hb1 = hb0 + (size_t)Bsz * Hh;
  unsigned short* emb = hb1 + (size_t)Bsz * Hh;
  float* decx = (float*)(emb + (size_t)Bsz * Hh);

  init_kernel<<<256, 256, 0, stream>>>(h, c, hb0, hb1);
  convert_kernel<<<4096, 256, 0, stream>>>(encWhh, decWih, decWhh,
                                           ebih, ebhh, dbih, dbhh,
                                           wEnc, wDecIh, wDecHh, biasEnc, biasDec);
  x0_kernel<<<1, 256, 0, stream>>>(x, lengths, decx);

  unsigned short* hbIn  = hb0;
  unsigned short* hbOut = hb1;
  for (int t = 0; t < Tlen; ++t) {
    lstm_step_kernel<<<Hh / 16, 512, 0, stream>>>(
        hbIn, wEnc, (const unsigned short*)nullptr, (const unsigned short*)nullptr,
        biasEnc, x, encWih, lengths, t, h, c, hbOut);
    unsigned short* tmp = hbIn; hbIn = hbOut; hbOut = tmp;
  }
  for (int s = 0; s < TGTn; ++s) {
    emb_kernel<<<512, 256, 0, stream>>>(decx, embW, embb, emb);
    lstm_step_kernel<<<Hh / 16, 512, 0, stream>>>(
        hbIn, wDecHh, emb, wDecIh,
        biasDec, (const float*)nullptr, (const float*)nullptr, (const int*)nullptr,
        0, h, c, hbOut);
    unsigned short* tmp = hbIn; hbIn = hbOut; hbOut = tmp;
    fc_kernel<<<32, 256, 0, stream>>>(h, fcW, fcb, outbuf, decx, s);
  }
  final_copy_kernel<<<256, 256, 0, stream>>>(h, c, outbuf);
}
